// GCNConvNet_7292854468800
// MI455X (gfx1250) — compile-verified, matrix operand order checked by reading, not run on hardware
//
#include <hip/hip_runtime.h>
#include <hip/hip_bf16.h>

typedef __bf16 bf16;
typedef __attribute__((ext_vector_type(16))) __bf16 v16bf;
typedef __attribute__((ext_vector_type(8)))  __bf16 v8bf;
typedef __attribute__((ext_vector_type(8)))  float   v8f;

static constexpr int NODES = 60000;
static constexpr int EDGES = 960000;
static constexpr int INC   = 128;
static constexpr int HIDC  = 512;
static constexpr int DCAT  = 1024;
static constexpr int DFC   = 2048;
static constexpr int DL1   = 4096;
static constexpr int OUTC  = 64;

enum { ACT_NONE = 0, ACT_RELU = 1, ACT_LEAKY = 2, ACT_SIG = 3 };

// ---------------- elementwise helpers ----------------

__global__ void copy_f32_kernel(const float* __restrict__ s, float* __restrict__ d, size_t n) {
    size_t i = (size_t)blockIdx.x * blockDim.x + threadIdx.x;
    if (i < n) d[i] = s[i];
}

__global__ void cvt_bf16_kernel(const float* __restrict__ s, bf16* __restrict__ d, size_t n) {
    size_t i = (size_t)blockIdx.x * blockDim.x + threadIdx.x;
    if (i < n) d[i] = (bf16)s[i];
}

// W[K,N] f32 (row-major) -> WT[N,K] bf16 (row-major), tiled through LDS so both
// global streams stay coalesced. 256 threads, 32x32 tile.
__global__ __launch_bounds__(256)
void cvt_transpose_kernel(const float* __restrict__ W, bf16* __restrict__ WT, int K, int N) {
    __shared__ float tile[32][33];
    const int kb = blockIdx.y * 32, nb = blockIdx.x * 32;
    const int tx = threadIdx.x & 31, ty = threadIdx.x >> 5;   // ty: 0..7
#pragma unroll
    for (int dy = 0; dy < 32; dy += 8) {
        int k = kb + ty + dy, n = nb + tx;
        tile[ty + dy][tx] = (k < K && n < N) ? W[(size_t)k * N + n] : 0.0f;
    }
    __syncthreads();
#pragma unroll
    for (int dy = 0; dy < 32; dy += 8) {
        int n = nb + ty + dy, k = kb + tx;
        if (n < N && k < K) WT[(size_t)n * K + k] = (bf16)tile[tx][ty + dy];
    }
}

// BN(eval) + linear bias folded into per-column scale/shift.
__global__ void bn_prep_kernel(const float* __restrict__ b, const float* __restrict__ gamma,
                               const float* __restrict__ beta, const float* __restrict__ mean,
                               const float* __restrict__ var,
                               float* __restrict__ scale, float* __restrict__ shift, int n) {
    int i = blockIdx.x * blockDim.x + threadIdx.x;
    if (i < n) {
        float s = gamma[i] * rsqrtf(var[i] + 1e-5f);
        scale[i] = s;
        shift[i] = b[i] * s + beta[i] - mean[i] * s;
    }
}

// GIN aggregation: h0 (pre-initialized to x) += sum over edges x[src] -> row dst.
// One wave per edge; fast f32 atomics, destination (30 MB) stays L2-resident.
__global__ void scatter_add_kernel(const float* __restrict__ x, const long long* __restrict__ ei,
                                   float* __restrict__ h0) {
    long long t = (long long)blockIdx.x * blockDim.x + threadIdx.x;
    int e    = (int)(t >> 5);
    int lane = (int)(t & 31);
    if (e >= EDGES) return;
    long long src = ei[e];
    long long dst = ei[(long long)EDGES + e];
    const float* xs = x  + (size_t)src * INC;
    float*       hd = h0 + (size_t)dst * INC;
#pragma unroll
    for (int c = 0; c < INC; c += 32)
        atomicAdd(&hd[c + lane], xs[c + lane]);
}

// ---------------- bf16 WMMA GEMM ----------------
// C[M,N] = act( (A[M,K] @ B[K,N]) * colScale + colShift ), A row pitch = lda,
// B passed pre-transposed as BT[N,K] bf16 row-major. Block tile 256x128 (8 waves,
// 4M x 2N), wave tile 64x64, K-step 32 -> 16 x v_wmma_f32_16x16x32_bf16 per wave
// per step. Next tile is prefetched into registers during the WMMA burst.
// Out-of-range rows/cols are handled by CLAMPING the load pointers (garbage only
// reaches C rows/cols that the store guards drop), so the hot loop is branchless.

template <typename OutT>
__global__ __launch_bounds__(256)
void gemm_bf16_kernel(const bf16* __restrict__ A, int lda,
                      const bf16* __restrict__ BT,
                      OutT* __restrict__ C,
                      const float* __restrict__ colScale,   // nullptr -> 1.0
                      const float* __restrict__ colShift,   // nullptr -> 0.0
                      int M, int N, int K, int ldc, int act) {
    constexpr int BM = 256, BN = 128, BK = 32;
    constexpr int PITCH = 40;   // 80B rows: 16B-aligned, 20-bank stride
    __shared__ alignas(16) bf16 As[BM * PITCH];
    __shared__ alignas(16) bf16 Bs[BN * PITCH];

    const int tid  = threadIdx.x;
    const int lane = tid & 31;
    const int wave = tid >> 5;
    const int wm   = wave & 3;    // 4 waves in M, 64 rows each
    const int wn   = wave >> 2;   // 2 waves in N, 64 cols each

    const int m0 = blockIdx.x * BM;
    const int n0 = blockIdx.y * BN;

    v8f acc[4][4];
#pragma unroll
    for (int i = 0; i < 4; ++i)
#pragma unroll
        for (int j = 0; j < 4; ++j)
#pragma unroll
            for (int e = 0; e < 8; ++e) acc[i][j][e] = 0.0f;

    // tile loader mapping: A -> one 256-row tile row per thread (32 cols);
    // B -> 128 rows x 2 threads/row (16 cols each). Pointers clamped, not masked.
    const int arow = tid;
    const int brow = tid & 127;
    const int bkh  = (tid >> 7) * 16;
    const int ga   = (m0 + arow < M) ? (m0 + arow) : (M - 1);
    const int gb   = (n0 + brow < N) ? (n0 + brow) : (N - 1);
    const bf16* aptr = A  + (size_t)ga * lda;
    const bf16* bptr = BT + (size_t)gb * K + bkh;

    v8bf ar[4], br[2];
    auto loadTiles = [&](int kk) {
#pragma unroll
        for (int q = 0; q < 4; ++q) ar[q] = *(const v8bf*)(aptr + kk + q * 8);
#pragma unroll
        for (int q = 0; q < 2; ++q) br[q] = *(const v8bf*)(bptr + kk + q * 8);
    };
    auto storeTiles = [&]() {
#pragma unroll
        for (int q = 0; q < 4; ++q) *(v8bf*)&As[arow * PITCH + q * 8] = ar[q];
#pragma unroll
        for (int q = 0; q < 2; ++q) *(v8bf*)&Bs[brow * PITCH + bkh + q * 8] = br[q];
    };

    const int rsel = lane & 15;
    const int kb   = (lane < 16) ? 0 : 8;

    loadTiles(0);
    for (int k0 = 0; k0 < K; k0 += BK) {
        __syncthreads();
        storeTiles();
        __syncthreads();
        if (k0 + BK < K) loadTiles(k0 + BK);   // overlaps the WMMA burst below

        v16bf bfrag[4];
#pragma unroll
        for (int j = 0; j < 4; ++j) {
            int r = wn * 64 + j * 16 + rsel;
            v8bf lo = *(const v8bf*)&Bs[r * PITCH + kb];
            v8bf hi = *(const v8bf*)&Bs[r * PITCH + kb + 16];
#pragma unroll
            for (int e = 0; e < 8; ++e) { bfrag[j][e] = lo[e]; bfrag[j][e + 8] = hi[e]; }
        }
#pragma unroll
        for (int i = 0; i < 4; ++i) {
            int r = wm * 64 + i * 16 + rsel;
            v8bf lo = *(const v8bf*)&As[r * PITCH + kb];
            v8bf hi = *(const v8bf*)&As[r * PITCH + kb + 16];
            v16bf afrag;
#pragma unroll
            for (int e = 0; e < 8; ++e) { afrag[e] = lo[e]; afrag[e + 8] = hi[e]; }
#pragma unroll
            for (int j = 0; j < 4; ++j)
                acc[i][j] = __builtin_amdgcn_wmma_f32_16x16x32_bf16(
                    false, afrag, false, bfrag[j], (short)0, acc[i][j], false, false);
        }
    }

    // epilogue: D layout -> lane covers col (lane&15), rows e + 8*(lane>=16)
    const int colLane = lane & 15;
    const int rowOff  = (lane >> 4) << 3;
#pragma unroll
    for (int j = 0; j < 4; ++j) {
        int col = n0 + wn * 64 + j * 16 + colLane;
        if (col >= N) continue;
        float s  = colScale ? colScale[col] : 1.0f;
        float sh = colShift ? colShift[col] : 0.0f;
#pragma unroll
        for (int i = 0; i < 4; ++i) {
            int rbase = m0 + wm * 64 + i * 16 + rowOff;
#pragma unroll
            for (int e = 0; e < 8; ++e) {
                int row = rbase + e;
                if (row >= M) continue;
                float y = acc[i][j][e] * s + sh;
                if (act == ACT_RELU)       y = y > 0.0f ? y : 0.0f;
                else if (act == ACT_LEAKY) y = y > 0.0f ? y : 0.01f * y;
                else if (act == ACT_SIG)   y = 1.0f / (1.0f + __expf(-y));
                C[(size_t)row * ldc + col] = (OutT)y;
            }
        }
    }
}

// ---------------- launch ----------------

extern "C" void kernel_launch(void* const* d_in, const int* in_sizes, int n_in,
                              void* d_out, int out_size, void* d_ws, size_t ws_size,
                              hipStream_t stream) {
    const float*     x  = (const float*)d_in[0];
    const long long* ei = (const long long*)d_in[1];
    const float* w1a = (const float*)d_in[2],  *b1a = (const float*)d_in[3];
    const float* ga  = (const float*)d_in[4],  *ba  = (const float*)d_in[5];
    const float* ma  = (const float*)d_in[6],  *va  = (const float*)d_in[7];
    const float* w2a = (const float*)d_in[8],  *b2a = (const float*)d_in[9];
    const float* w1b = (const float*)d_in[10], *b1b = (const float*)d_in[11];
    const float* gb  = (const float*)d_in[12], *bb  = (const float*)d_in[13];
    const float* mb  = (const float*)d_in[14], *vb  = (const float*)d_in[15];
    const float* w2b = (const float*)d_in[16], *b2b = (const float*)d_in[17];
    const float* fcw = (const float*)d_in[18], *fcb = (const float*)d_in[19];
    const float* l1w = (const float*)d_in[20], *l1b = (const float*)d_in[21];
    const float* l2w = (const float*)d_in[22], *l2b = (const float*)d_in[23];
    const float* ow  = (const float*)d_in[24], *ob  = (const float*)d_in[25];

    // ---- carve workspace ----
    char* ws = (char*)d_ws;
    size_t off = 0;
    auto carve = [&](size_t bytes) -> void* {
        void* p = ws + off;
        off = (off + bytes + 255) & ~(size_t)255;
        return p;
    };
    // fused layer-1 weight: rows 0..511 = w1a^T, rows 512..1023 = w1b^T  ([N,K], K=128)
    bf16* w1ab_t = (bf16*)carve((size_t)DCAT * INC * 2);
    bf16* w2a_t  = (bf16*)carve((size_t)HIDC * HIDC * 2);
    bf16* w2b_t  = (bf16*)carve((size_t)HIDC * HIDC * 2);
    bf16* fcw_t  = (bf16*)carve((size_t)DCAT * DFC  * 2);
    bf16* l1w_t  = (bf16*)carve((size_t)DFC  * DL1  * 2);
    bf16* l2w_t  = (bf16*)carve((size_t)DL1  * DFC  * 2);
    bf16* ow_t   = (bf16*)carve((size_t)DFC  * OUTC * 2);
    float* sc_ab = (float*)carve(DCAT * 4);   // [scale_a | scale_b]
    float* sh_ab = (float*)carve(DCAT * 4);   // [shift_a | shift_b]
    float* h0    = (float*)carve((size_t)NODES * INC * 4);
    bf16*  h0h   = (bf16*)carve((size_t)NODES * INC * 2);
    bf16*  h1    = (bf16*)carve((size_t)NODES * DCAT * 2);  // fused [H1a | H1b]
    bf16*  hcat  = (bf16*)carve((size_t)NODES * DCAT * 2);  // [x1 | x2]
    bf16*  ebuf  = (bf16*)carve((size_t)NODES * DFC * 2);   // F1, later F3
    bf16*  fbuf  = (bf16*)carve((size_t)NODES * DL1 * 2);   // F2
    (void)ws_size; (void)n_in; (void)in_sizes; (void)out_size;

    auto cvtT = [&](const float* W, bf16* WT, int K, int N) {
        dim3 g((N + 31) / 32, (K + 31) / 32);
        cvt_transpose_kernel<<<g, 256, 0, stream>>>(W, WT, K, N);
    };

    // weights -> transposed bf16 (layer-1 branches concatenated along N)
    cvtT(w1a, w1ab_t,                      INC,  HIDC);
    cvtT(w1b, w1ab_t + (size_t)HIDC * INC, INC,  HIDC);
    cvtT(w2a, w2a_t, HIDC, HIDC);
    cvtT(w2b, w2b_t, HIDC, HIDC);
    cvtT(fcw, fcw_t, DCAT, DFC);
    cvtT(l1w, l1w_t, DFC,  DL1);
    cvtT(l2w, l2w_t, DL1,  DFC);
    cvtT(ow,  ow_t,  DFC,  OUTC);

    // fold BN into concatenated scale/shift
    bn_prep_kernel<<<(HIDC + 255) / 256, 256, 0, stream>>>(b1a, ga, ba, ma, va, sc_ab, sh_ab, HIDC);
    bn_prep_kernel<<<(HIDC + 255) / 256, 256, 0, stream>>>(b1b, gb, bb, mb, vb,
                                                           sc_ab + HIDC, sh_ab + HIDC, HIDC);

    // h0 = x + segment_sum(x[src] -> dst)
    {
        size_t n = (size_t)NODES * INC;
        copy_f32_kernel<<<(unsigned)((n + 255) / 256), 256, 0, stream>>>(x, h0, n);
        long long tot = (long long)EDGES * 32;
        scatter_add_kernel<<<(unsigned)((tot + 255) / 256), 256, 0, stream>>>(x, ei, h0);
        cvt_bf16_kernel<<<(unsigned)((n + 255) / 256), 256, 0, stream>>>(h0, h0h, n);
    }

    const int M = NODES;
    dim3 blk(256);
    auto gdim = [](int m, int n) { return dim3((m + 255) / 256, (n + 127) / 128); };

    // GIN layer 1 (both branches fused): relu(BN(h0 @ [w1a|w1b] + [b1a|b1b]))
    gemm_bf16_kernel<bf16><<<gdim(M, DCAT), blk, 0, stream>>>(
        h0h, INC, w1ab_t, h1, sc_ab, sh_ab, M, DCAT, INC, DCAT, ACT_RELU);
    // GIN layer 2, branch A: relu(H1a @ w2a + b2a) -> hcat[:, 0:512]
    gemm_bf16_kernel<bf16><<<gdim(M, HIDC), blk, 0, stream>>>(
        h1, DCAT, w2a_t, hcat, nullptr, b2a, M, HIDC, HIDC, DCAT, ACT_RELU);
    // GIN layer 2, branch B: relu(H1b @ w2b + b2b) -> hcat[:, 512:1024]
    gemm_bf16_kernel<bf16><<<gdim(M, HIDC), blk, 0, stream>>>(
        h1 + HIDC, DCAT, w2b_t, hcat + HIDC, nullptr, b2b, M, HIDC, HIDC, DCAT, ACT_RELU);

    // head
    gemm_bf16_kernel<bf16><<<gdim(M, DFC), blk, 0, stream>>>(
        hcat, DCAT, fcw_t, ebuf, nullptr, fcb, M, DFC, DCAT, DFC, ACT_LEAKY);
    gemm_bf16_kernel<bf16><<<gdim(M, DL1), blk, 0, stream>>>(
        ebuf, DFC, l1w_t, fbuf, nullptr, l1b, M, DL1, DFC, DL1, ACT_NONE);
    gemm_bf16_kernel<bf16><<<gdim(M, DFC), blk, 0, stream>>>(
        fbuf, DL1, l2w_t, ebuf, nullptr, l2b, M, DFC, DL1, DFC, ACT_NONE);
    gemm_bf16_kernel<float><<<gdim(M, OUTC), blk, 0, stream>>>(
        ebuf, DFC, ow_t, (float*)d_out, nullptr, ob, M, OUTC, DFC, OUTC, ACT_SIG);
}
